// AttentionPairBias_10445360464027
// MI455X (gfx1250) — matrix operand order repaired and checked by reading, not compiled, not run on hardware
//
#include <hip/hip_runtime.h>
#include <hip/hip_bf16.h>

// ---------------------------------------------------------------------------
// Types
// ---------------------------------------------------------------------------
typedef __bf16 bf16;
typedef __attribute__((ext_vector_type(16))) __bf16 v16bf;
typedef __attribute__((ext_vector_type(8)))  __bf16 bf16x8;
typedef __attribute__((ext_vector_type(8)))  float  v8f;

#define N_SEQ 1024
#define CQ    768
#define CZ    128
#define NH    16
#define DH    48
#define HD    768          // NH*DH
#define DHP   64           // DH padded to WMMA K granularity
#define EPSLN 1e-5f
#define INFB  1.0e9f

// ---------------------------------------------------------------------------
// bf16 helpers (round-to-nearest-even via bits; avoids relying on f32<->bf16
// implicit conversion support)
// ---------------------------------------------------------------------------
__device__ __forceinline__ bf16 f2bf(float f) {
  unsigned u; __builtin_memcpy(&u, &f, 4);
  unsigned r = u + 0x7fffu + ((u >> 16) & 1u);
  unsigned short h = (unsigned short)(r >> 16);
  bf16 o; __builtin_memcpy(&o, &h, 2);
  return o;
}
__device__ __forceinline__ float bf2f(bf16 x) {
  unsigned short h; __builtin_memcpy(&h, &x, 2);
  unsigned u = ((unsigned)h) << 16;
  float f; __builtin_memcpy(&f, &u, 4);
  return f;
}

// ---------------------------------------------------------------------------
// WMMA fragment loaders (ISA 16-bit layouts, wave32)
// A 16x32 (MxK): lane l holds row m=l&15; element e <-> K = e + (l>=16)*8 + (e>=8)*8
//   -> two contiguous 8-elem (16B) loads at  base + half*8  and  base + half*8 + 16
// B 32x16 (KxN): lane l holds col n=l&15; element e <-> K = e + (l>=16)*8 + (e>=8)*8
//   -> two contiguous 8-elem (16B) loads at  base + half*8  and  base + half*8 + 16
// ---------------------------------------------------------------------------
__device__ __forceinline__ v16bf load_frag_a(const bf16* base, int half) {
  bf16x8 lo = *(const bf16x8*)(base + half * 8);
  bf16x8 hi = *(const bf16x8*)(base + half * 8 + 16);
  v16bf r;
#pragma unroll
  for (int i = 0; i < 8; ++i) { r[i] = lo[i]; r[i + 8] = hi[i]; }
  return r;
}
__device__ __forceinline__ v16bf load_frag_b(const bf16* base, int half) {
  bf16x8 lo = *(const bf16x8*)(base + half * 8);
  bf16x8 hi = *(const bf16x8*)(base + half * 8 + 16);
  v16bf r;
#pragma unroll
  for (int i = 0; i < 8; ++i) { r[i] = lo[i]; r[i + 8] = hi[i]; }
  return r;
}

// half-wave (16-lane) reductions; C-layout rows live within one 16-lane half
__device__ __forceinline__ float hmax16(float v) {
  v = fmaxf(v, __shfl_xor(v, 1, 16));
  v = fmaxf(v, __shfl_xor(v, 2, 16));
  v = fmaxf(v, __shfl_xor(v, 4, 16));
  v = fmaxf(v, __shfl_xor(v, 8, 16));
  return v;
}
__device__ __forceinline__ float hsum16(float v) {
  v += __shfl_xor(v, 1, 16);
  v += __shfl_xor(v, 2, 16);
  v += __shfl_xor(v, 4, 16);
  v += __shfl_xor(v, 8, 16);
  return v;
}

// ---------------------------------------------------------------------------
// Kernel 1: transpose + convert f32 [K][Nc] -> bf16 [Nc][K]
// ---------------------------------------------------------------------------
__global__ void k_transpose_bf16(const float* __restrict__ in, bf16* __restrict__ out,
                                 int K, int Nc) {
  int id = blockIdx.x * 256 + threadIdx.x;
  if (id >= K * Nc) return;
  int n = id / K, k = id - n * K;
  out[(size_t)n * K + k] = f2bf(in[(size_t)k * Nc + n]);
}

// ---------------------------------------------------------------------------
// Kernel 2: LayerNorm(a) -> bf16 [N][CQ]
// ---------------------------------------------------------------------------
__global__ void k_ln_a(const float* __restrict__ a, const float* __restrict__ w,
                       const float* __restrict__ b, bf16* __restrict__ out) {
  int row = blockIdx.x;
  int t = threadIdx.x;                       // 256 threads
  const float* x = a + (size_t)row * CQ;
  float v[3], s = 0.f, sq = 0.f;
#pragma unroll
  for (int i = 0; i < 3; ++i) { v[i] = x[t + i * 256]; s += v[i]; sq += v[i] * v[i]; }
  // wave reduce
#pragma unroll
  for (int m = 1; m < 32; m <<= 1) { s += __shfl_xor(s, m, 32); sq += __shfl_xor(sq, m, 32); }
  __shared__ float red[2][8];
  if ((t & 31) == 0) { red[0][t >> 5] = s; red[1][t >> 5] = sq; }
  __syncthreads();
  float ts = 0.f, tq = 0.f;
#pragma unroll
  for (int wv = 0; wv < 8; ++wv) { ts += red[0][wv]; tq += red[1][wv]; }
  float mean = ts * (1.f / CQ);
  float var = tq * (1.f / CQ) - mean * mean;
  float rstd = rsqrtf(var + EPSLN);
#pragma unroll
  for (int i = 0; i < 3; ++i) {
    int c = t + i * 256;
    out[(size_t)row * CQ + c] = f2bf((v[i] - mean) * rstd * w[c] + b[c]);
  }
}

// ---------------------------------------------------------------------------
// Kernel 3: fused LayerNorm(z) + pair bias GEMM (z_ln @ wz) -> pb[h][i][j] bf16
// One 128-thread block per (i, 16-j tile). Wave 0 runs 4x v_wmma (K=128).
// ---------------------------------------------------------------------------
__global__ void k_pair_bias(const float* __restrict__ z, const float* __restrict__ lnw,
                            const float* __restrict__ lnb, const bf16* __restrict__ wzT,
                            bf16* __restrict__ pb) {
  int bid = blockIdx.x;
  int i  = bid >> 6;
  int j0 = (bid & 63) << 4;
  int t  = threadIdx.x;                      // 0..127
  __shared__ bf16  zt[16][CZ];               // normalized tile (bf16)
  __shared__ float cb[16][17];               // C transpose staging (jj, h)

  int jj  = t >> 3;                          // row within tile (0..15)
  int cb8 = (t & 7) * 16;                    // channel base for this thread
  const float* zr = z + (((size_t)i * N_SEQ) + (j0 + jj)) * CZ + cb8;
  __builtin_prefetch(zr + 16 * CZ, 0, 0);    // next j-tile of same row i

  float xv[16];
#pragma unroll
  for (int q4 = 0; q4 < 4; ++q4) {
    float4 f = ((const float4*)zr)[q4];
    xv[q4 * 4 + 0] = f.x; xv[q4 * 4 + 1] = f.y; xv[q4 * 4 + 2] = f.z; xv[q4 * 4 + 3] = f.w;
  }
  float s = 0.f, sq = 0.f;
#pragma unroll
  for (int e = 0; e < 16; ++e) { s += xv[e]; sq += xv[e] * xv[e]; }
  // reduce across the 8 threads owning this row (aligned 8-lane groups)
#pragma unroll
  for (int m = 1; m < 8; m <<= 1) { s += __shfl_xor(s, m, 8); sq += __shfl_xor(sq, m, 8); }
  float mean = s * (1.f / CZ);
  float var  = sq * (1.f / CZ) - mean * mean;
  float rstd = rsqrtf(var + EPSLN);
#pragma unroll
  for (int e = 0; e < 16; ++e) {
    int c = cb8 + e;
    zt[jj][c] = f2bf((xv[e] - mean) * rstd * lnw[c] + lnb[c]);
  }
  __syncthreads();

  if (t < 32) {                               // wave 0: 16(j) x 16(h) tile, K=128
    int half = t >> 4, ln = t & 15;
    v8f c = {};
#pragma unroll
    for (int kc = 0; kc < CZ; kc += 32) {
      v16bf af = load_frag_a(&zt[ln][kc], half);
      v16bf bfrag = load_frag_b(wzT + (size_t)ln * CZ + kc, half);
      c = __builtin_amdgcn_wmma_f32_16x16x32_bf16(false, af, false, bfrag,
                                                  (short)0, c, false, false);
    }
#pragma unroll
    for (int r = 0; r < 8; ++r) cb[r + half * 8][ln] = c[r];
  }
  __syncthreads();

  // coalesced store: pb[h][i][j0..j0+15]
  int h = t >> 3, j2 = (t & 7) * 2;
  size_t base = (((size_t)h * N_SEQ) + i) * N_SEQ + j0;
  pb[base + j2 + 0] = f2bf(cb[j2 + 0][h]);
  pb[base + j2 + 1] = f2bf(cb[j2 + 1][h]);
}

// ---------------------------------------------------------------------------
// Kernel 4: generic bf16 WMMA GEMM  C[M][Nc] = epilogue(A[M][K] @ Bt[Nc][K]^T)
// One wave per 16x16 tile; 8 waves per block.
// mode 0: C = acc*scale   mode 1: C = sigmoid(acc + bias[n])   mode 2: C = acc + bias[n]
// ---------------------------------------------------------------------------
__global__ void k_gemm_bf16(const bf16* __restrict__ A, const bf16* __restrict__ Bt,
                            float* __restrict__ C, int M, int Nc, int K,
                            int mode, float scale, const float* __restrict__ bias) {
  int wave = threadIdx.x >> 5;
  int l    = threadIdx.x & 31;
  int tile = blockIdx.x * 8 + wave;
  int ntn  = Nc >> 4;
  int tm   = tile / ntn, tn = tile - tm * ntn;
  if (tm >= (M >> 4)) return;                // wave-uniform
  int half = l >> 4, ln = l & 15;

  const bf16* arow = A  + (size_t)(tm * 16 + ln) * K;
  const bf16* brow = Bt + (size_t)(tn * 16 + ln) * K;
  v8f c = {};
  for (int kc = 0; kc < K; kc += 32) {
    v16bf af = load_frag_a(arow + kc, half);
    v16bf bf_ = load_frag_b(brow + kc, half);
    c = __builtin_amdgcn_wmma_f32_16x16x32_bf16(false, af, false, bf_,
                                                (short)0, c, false, false);
  }
#pragma unroll
  for (int r = 0; r < 8; ++r) {
    int m = tm * 16 + r + half * 8;
    int n = tn * 16 + ln;
    float v = c[r];
    if (mode == 0)      v *= scale;
    else if (mode == 1) v = 1.f / (1.f + __expf(-(v + bias[n])));
    else                v += bias[n];
    C[(size_t)m * Nc + n] = v;
  }
}

// ---------------------------------------------------------------------------
// Kernel 5: pack q/k into per-head [h][n][64] bf16 (DH padded 48->64 w/ zeros)
// ---------------------------------------------------------------------------
__global__ void k_pack_qk(const float* __restrict__ in, bf16* __restrict__ out) {
  int id = blockIdx.x * 256 + threadIdx.x;   // over NH*N_SEQ*DHP
  if (id >= NH * N_SEQ * DHP) return;
  int h = id / (N_SEQ * DHP);
  int r = id - h * (N_SEQ * DHP);
  int n = r / DHP, d = r - n * DHP;
  bf16 v; unsigned short zz = 0; __builtin_memcpy(&v, &zz, 2);
  if (d < DH) v = f2bf(in[(size_t)n * HD + h * DH + d]);
  out[id] = v;
}

// Kernel 6: v transpose -> [h][d][n] bf16
__global__ void k_pack_v(const float* __restrict__ in, bf16* __restrict__ out) {
  int id = blockIdx.x * 256 + threadIdx.x;   // over NH*DH*N_SEQ
  if (id >= NH * DH * N_SEQ) return;
  int h = id / (DH * N_SEQ);
  int r = id - h * (DH * N_SEQ);
  int d = r / N_SEQ, n = r - d * N_SEQ;
  out[id] = f2bf(in[(size_t)n * HD + h * DH + d]);
}

// ---------------------------------------------------------------------------
// Kernel 7: flash attention. One wave per (head, 16-query block).
// logits via WMMA (K=64 padded), online softmax, P@V via WMMA (3 dim-tiles).
// Gate + 1/l applied in epilogue; writes bf16 o for final projection.
// ---------------------------------------------------------------------------
__global__ void __launch_bounds__(32)
k_attn(const bf16* __restrict__ qp, const bf16* __restrict__ kp,
       const bf16* __restrict__ vt, const bf16* __restrict__ pb,
       const float* __restrict__ mask, const float* __restrict__ g,
       bf16* __restrict__ obf) {
  int h  = blockIdx.x >> 6;
  int qb = (blockIdx.x & 63) << 4;
  int l  = threadIdx.x;
  int half = l >> 4, ln = l & 15;
  __shared__ bf16 P[16][32];

  const bf16* qrow = qp + (((size_t)h * N_SEQ) + qb + ln) * DHP;
  v16bf a0 = load_frag_a(qrow + 0, half);
  v16bf a1 = load_frag_a(qrow + 32, half);

  v8f o0 = {}, o1 = {}, o2 = {};
  float mx[8], sm[8];
#pragma unroll
  for (int r = 0; r < 8; ++r) { mx[r] = -1e30f; sm[r] = 0.f; }

  for (int jc = 0; jc < N_SEQ; jc += 32) {
    // ---- logits: two 16-key subtiles, K = 64 (2 WMMA each)
    v8f s0 = {}, s1 = {};
    {
      const bf16* krow = kp + (((size_t)h * N_SEQ) + jc + ln) * DHP;
      v16bf b0 = load_frag_b(krow + 0, half);
      v16bf b1 = load_frag_b(krow + 32, half);
      s0 = __builtin_amdgcn_wmma_f32_16x16x32_bf16(false, a0, false, b0, (short)0, s0, false, false);
      s0 = __builtin_amdgcn_wmma_f32_16x16x32_bf16(false, a1, false, b1, (short)0, s0, false, false);
    }
    {
      const bf16* krow = kp + (((size_t)h * N_SEQ) + jc + 16 + ln) * DHP;
      v16bf b0 = load_frag_b(krow + 0, half);
      v16bf b1 = load_frag_b(krow + 32, half);
      s1 = __builtin_amdgcn_wmma_f32_16x16x32_bf16(false, a0, false, b0, (short)0, s1, false, false);
      s1 = __builtin_amdgcn_wmma_f32_16x16x32_bf16(false, a1, false, b1, (short)0, s1, false, false);
    }
    // ---- add pair bias + mask bias
    float mb0 = INFB * (mask[jc + ln] - 1.f);
    float mb1 = INFB * (mask[jc + 16 + ln] - 1.f);
#pragma unroll
    for (int r = 0; r < 8; ++r) {
      int qi = qb + r + half * 8;
      const bf16* pbr = pb + (((size_t)h * N_SEQ) + qi) * N_SEQ + jc;
      s0[r] += bf2f(pbr[ln]) + mb0;
      s1[r] += bf2f(pbr[16 + ln]) + mb1;
    }
    // ---- online softmax (rows live within a 16-lane half)
#pragma unroll
    for (int r = 0; r < 8; ++r) {
      float cm = hmax16(fmaxf(s0[r], s1[r]));
      float nm = fmaxf(mx[r], cm);
      float al = __expf(mx[r] - nm);
      mx[r] = nm;
      float p0 = __expf(s0[r] - nm);
      float p1 = __expf(s1[r] - nm);
      sm[r] = sm[r] * al + hsum16(p0 + p1);
      o0[r] *= al; o1[r] *= al; o2[r] *= al;
      P[r + half * 8][ln]      = f2bf(p0);
      P[r + half * 8][16 + ln] = f2bf(p1);
    }
    __syncthreads();
    // ---- O += P @ V
    v16bf pa = load_frag_a(&P[ln][0], half);
    {
      const bf16* vrow = vt + (((size_t)h * DH) + 0 + ln) * N_SEQ + jc;
      v16bf vb = load_frag_b(vrow, half);
      o0 = __builtin_amdgcn_wmma_f32_16x16x32_bf16(false, pa, false, vb, (short)0, o0, false, false);
    }
    {
      const bf16* vrow = vt + (((size_t)h * DH) + 16 + ln) * N_SEQ + jc;
      v16bf vb = load_frag_b(vrow, half);
      o1 = __builtin_amdgcn_wmma_f32_16x16x32_bf16(false, pa, false, vb, (short)0, o1, false, false);
    }
    {
      const bf16* vrow = vt + (((size_t)h * DH) + 32 + ln) * N_SEQ + jc;
      v16bf vb = load_frag_b(vrow, half);
      o2 = __builtin_amdgcn_wmma_f32_16x16x32_bf16(false, pa, false, vb, (short)0, o2, false, false);
    }
    __syncthreads();   // protect P (WAR) before next iteration
  }
  // ---- epilogue: 1/l, gate, write bf16
#pragma unroll
  for (int r = 0; r < 8; ++r) {
    float inv = 1.f / sm[r];
    int qi = qb + r + half * 8;
    size_t base = (size_t)qi * HD + h * DH;
    float g0 = g[base + 0 + ln], g1 = g[base + 16 + ln], g2 = g[base + 32 + ln];
    obf[base + 0 + ln]  = f2bf(o0[r] * inv * g0);
    obf[base + 16 + ln] = f2bf(o1[r] * inv * g1);
    obf[base + 32 + ln] = f2bf(o2[r] * inv * g2);
  }
}

// ---------------------------------------------------------------------------
// Host launcher
// ---------------------------------------------------------------------------
extern "C" void kernel_launch(void* const* d_in, const int* in_sizes, int n_in,
                              void* d_out, int out_size, void* d_ws, size_t ws_size,
                              hipStream_t stream) {
  const float* a     = (const float*)d_in[0];
  const float* z     = (const float*)d_in[1];
  const float* mask  = (const float*)d_in[2];
  const float* ln_aw = (const float*)d_in[3];
  const float* ln_ab = (const float*)d_in[4];
  const float* ln_zw = (const float*)d_in[5];
  const float* ln_zb = (const float*)d_in[6];
  const float* wz    = (const float*)d_in[7];
  const float* wq    = (const float*)d_in[8];
  const float* wk    = (const float*)d_in[9];
  const float* wv    = (const float*)d_in[10];
  const float* wg    = (const float*)d_in[11];
  const float* bg    = (const float*)d_in[12];
  const float* wo    = (const float*)d_in[13];
  const float* bo    = (const float*)d_in[14];
  float* out = (float*)d_out;

  char* w = (char*)d_ws;
  size_t off = 0;
  auto alloc = [&](size_t bytes) -> void* {
    void* p = w + off;
    off += (bytes + 255) & ~(size_t)255;
    return p;
  };
  bf16* alnb = (bf16*)alloc((size_t)N_SEQ * CQ * 2);
  bf16* wqT  = (bf16*)alloc((size_t)HD * CQ * 2);
  bf16* wkT  = (bf16*)alloc((size_t)HD * CQ * 2);
  bf16* wvT  = (bf16*)alloc((size_t)HD * CQ * 2);
  bf16* wgT  = (bf16*)alloc((size_t)HD * CQ * 2);
  bf16* woT  = (bf16*)alloc((size_t)CQ * HD * 2);
  bf16* wzT  = (bf16*)alloc((size_t)NH * CZ * 2);
  float* qf  = (float*)alloc((size_t)N_SEQ * HD * 4);
  float* kf  = (float*)alloc((size_t)N_SEQ * HD * 4);
  float* vf  = (float*)alloc((size_t)N_SEQ * HD * 4);
  float* gf  = (float*)alloc((size_t)N_SEQ * HD * 4);
  bf16* qpk  = (bf16*)alloc((size_t)NH * N_SEQ * DHP * 2);
  bf16* kpk  = (bf16*)alloc((size_t)NH * N_SEQ * DHP * 2);
  bf16* vtp  = (bf16*)alloc((size_t)NH * DH * N_SEQ * 2);
  bf16* pbb  = (bf16*)alloc((size_t)NH * N_SEQ * N_SEQ * 2);
  bf16* obf  = (bf16*)alloc((size_t)N_SEQ * HD * 2);
  (void)ws_size; (void)in_sizes; (void)n_in; (void)out_size;

  const int TB = 256;
  // weight transposes -> bf16 [Nc][K]
  int ew = CQ * HD;
  k_transpose_bf16<<<(ew + TB - 1) / TB, TB, 0, stream>>>(wq, wqT, CQ, HD);
  k_transpose_bf16<<<(ew + TB - 1) / TB, TB, 0, stream>>>(wk, wkT, CQ, HD);
  k_transpose_bf16<<<(ew + TB - 1) / TB, TB, 0, stream>>>(wv, wvT, CQ, HD);
  k_transpose_bf16<<<(ew + TB - 1) / TB, TB, 0, stream>>>(wg, wgT, CQ, HD);
  k_transpose_bf16<<<(ew + TB - 1) / TB, TB, 0, stream>>>(wo, woT, HD, CQ);
  k_transpose_bf16<<<(CZ * NH + TB - 1) / TB, TB, 0, stream>>>(wz, wzT, CZ, NH);

  // LN(a)
  k_ln_a<<<N_SEQ, 256, 0, stream>>>(a, ln_aw, ln_ab, alnb);

  // fused LN(z) + pair bias (the memory-bound hot loop: 537 MB of z, read once)
  k_pair_bias<<<N_SEQ * (N_SEQ / 16), 128, 0, stream>>>(z, ln_zw, ln_zb, wzT, pbb);

  // projections: q (scaled), k, v, g (sigmoid gate)
  int gblocks = ((N_SEQ / 16) * (HD / 16)) / 8;   // 384
  const float qscale = 0.14433756729740643f;      // 1/sqrt(48)
  k_gemm_bf16<<<gblocks, 256, 0, stream>>>(alnb, wqT, qf, N_SEQ, HD, CQ, 0, qscale, nullptr);
  k_gemm_bf16<<<gblocks, 256, 0, stream>>>(alnb, wkT, kf, N_SEQ, HD, CQ, 0, 1.f, nullptr);
  k_gemm_bf16<<<gblocks, 256, 0, stream>>>(alnb, wvT, vf, N_SEQ, HD, CQ, 0, 1.f, nullptr);
  k_gemm_bf16<<<gblocks, 256, 0, stream>>>(alnb, wgT, gf, N_SEQ, HD, CQ, 1, 1.f, bg);

  // repack for attention fragments
  int eqk = NH * N_SEQ * DHP;
  k_pack_qk<<<(eqk + TB - 1) / TB, TB, 0, stream>>>(qf, qpk);
  k_pack_qk<<<(eqk + TB - 1) / TB, TB, 0, stream>>>(kf, kpk);
  int ev = NH * DH * N_SEQ;
  k_pack_v<<<(ev + TB - 1) / TB, TB, 0, stream>>>(vf, vtp);

  // attention (one wave per head x 16-query block)
  k_attn<<<NH * (N_SEQ / 16), 32, 0, stream>>>(qpk, kpk, vtp, pbb, mask, gf, obf);

  // output projection + bias
  k_gemm_bf16<<<gblocks, 256, 0, stream>>>(obf, woT, out, N_SEQ, CQ, HD, 2, 1.f, bo);
}
